// LocalTransformer_77171972374825
// MI455X (gfx1250) — matrix operand order repaired and checked by reading, not compiled, not run on hardware
//
#include <hip/hip_runtime.h>
#include <math.h>
#include <stdint.h>

// ---------------------------------------------------------------------------
// CDNA5 (gfx1250) wave32 WMMA implementation of the local-attention
// transformer. f16 storage for GEMM operands, f32 accumulation via
// v_wmma_f32_16x16x32_f16. Weights are pre-cast/pre-transposed to f16 with
// zero padding so the steady-state GEMMs are branch-free. GEMM tiles are
// staged with GLOBAL_LOAD_ASYNC_TO_LDS_B128 (ASYNCcnt-tracked DMA, no VGPR
// round trip).
// ---------------------------------------------------------------------------

typedef _Float16 f16;
typedef __attribute__((ext_vector_type(16))) _Float16 v16h;
typedef __attribute__((ext_vector_type(8)))  _Float16 v8h;
typedef __attribute__((ext_vector_type(8)))  float    v8f;

#define D_MODEL 512
#define SEQ     2048
#define NTOK    16384      // B * N
#define NHEAD   8
#define DHD     64
#define WSZ     128
#define NWIN    16
#define FF1     1365
#define FF2     2730
#define FFP     1376       // FF1 padded to multiple of 32 (K pad for ff2)
#define FF2P    2816       // 2*FF1 padded to multiple of 128 (N pad for ff1)
#define QKVN    1536

__device__ __forceinline__ v16h ld16(const f16* p0, const f16* p1) {
  v8h a = *(const v8h*)p0;
  v8h b = *(const v8h*)p1;
  v16h r;
#pragma unroll
  for (int i = 0; i < 8; ++i) { r[i] = a[i]; r[i + 8] = b[i]; }
  return r;
}

__device__ __forceinline__ v8f wmma16(v16h a, v16h b, v8f c) {
  return __builtin_amdgcn_wmma_f32_16x16x32_f16(false, a, false, b,
                                                (short)0, c, false, false);
}

// Async DMA of 64 contiguous bytes (per lane) from global to LDS.
// INST_OFFSET is added to both the LDS and global addresses (ISA 08 §4.4).
__device__ __forceinline__ void async_copy64(unsigned lds_addr,
                                             unsigned long long gaddr) {
  asm volatile(
      "global_load_async_to_lds_b128 %0, %1, off\n\t"
      "global_load_async_to_lds_b128 %0, %1, off offset:16\n\t"
      "global_load_async_to_lds_b128 %0, %1, off offset:32\n\t"
      "global_load_async_to_lds_b128 %0, %1, off offset:48"
      :: "v"(lds_addr), "v"(gaddr) : "memory");
}

__device__ __forceinline__ void async_wait0() {
  asm volatile("s_wait_asynccnt 0" ::: "memory");
}

// ---------------------------------------------------------------------------
// Weight cast + transpose: W[K][N] f32 -> Wt[Npad][Kpad] f16 (zero padded)
// ---------------------------------------------------------------------------
__global__ __launch_bounds__(256) void wcast_kernel(
    const float* __restrict__ W, f16* __restrict__ Wt,
    int K, int N, int Kpad, int Npad) {
  size_t idx = (size_t)blockIdx.x * 256 + threadIdx.x;
  if (idx >= (size_t)Npad * Kpad) return;
  int n = (int)(idx / Kpad);
  int k = (int)(idx - (size_t)n * Kpad);
  float v = (n < N && k < K) ? W[(size_t)k * N + n] : 0.f;
  Wt[idx] = (f16)v;
}

// ---------------------------------------------------------------------------
// Embedding + sinusoidal position encoding + padding mask
// ---------------------------------------------------------------------------
__global__ __launch_bounds__(256) void embed_kernel(
    const int* __restrict__ x, const float* __restrict__ emb,
    float* __restrict__ hout, float* __restrict__ maskf) {
  int row = blockIdx.x;            // b*N + n
  int n = row & (SEQ - 1);
  int t = threadIdx.x;
  int tok = x[row];
  float mk = (tok != 0) ? 1.f : 0.f;
  if (t == 0) maskf[row] = mk;
  const float c = 9.210340371976184f / 256.f;   // ln(10000)/256
#pragma unroll
  for (int d = t; d < D_MODEL; d += 256) {
    int k = d & 255;
    float inv = expf(-(float)k * c);
    float sarg = (float)n * inv;
    float pe = (d < 256) ? sinf(sarg) : cosf(sarg);
    hout[(size_t)row * D_MODEL + d] = emb[(size_t)tok * D_MODEL + d] * mk + pe;
  }
}

// ---------------------------------------------------------------------------
// LayerNorm (row of 512) -> f16 cast
// ---------------------------------------------------------------------------
__global__ __launch_bounds__(256) void ln_cast_kernel(
    const float* __restrict__ hin, const float* __restrict__ g,
    const float* __restrict__ bb, f16* __restrict__ z) {
  __shared__ float red[16];
  int row = blockIdx.x, t = threadIdx.x;
  float x0 = hin[(size_t)row * D_MODEL + t];
  float x1 = hin[(size_t)row * D_MODEL + 256 + t];
  float s = x0 + x1, q = x0 * x0 + x1 * x1;
#pragma unroll
  for (int m = 1; m < 32; m <<= 1) {
    s += __shfl_xor(s, m, 32);
    q += __shfl_xor(q, m, 32);
  }
  int wave = t >> 5, lane = t & 31;
  if (lane == 0) { red[wave] = s; red[8 + wave] = q; }
  __syncthreads();
  if (t == 0) {
    float S = 0.f, Q = 0.f;
    for (int i = 0; i < 8; i++) { S += red[i]; Q += red[8 + i]; }
    red[0] = S; red[1] = Q;
  }
  __syncthreads();
  float mean = red[0] * (1.f / 512.f);
  float var = red[1] * (1.f / 512.f) - mean * mean;
  float rs = rsqrtf(var + 1e-5f);
  z[(size_t)row * D_MODEL + t]       = (f16)((x0 - mean) * rs * g[t] + bb[t]);
  z[(size_t)row * D_MODEL + 256 + t] = (f16)((x1 - mean) * rs * g[256 + t] + bb[256 + t]);
}

// ---------------------------------------------------------------------------
// Dynamic position bias MLP (tiny)
// ---------------------------------------------------------------------------
__global__ void dpb1_kernel(const float* __restrict__ w1,
                            const float* __restrict__ b1,
                            float* __restrict__ h1) {
  int idx = blockIdx.x * 256 + threadIdx.x;     // 65536 = 256 x 256
  int r = idx >> 8, c = idx & 255;
  float v = (float)r * w1[c] + b1[c];
  h1[idx] = v / (1.f + expf(-v));
}

__global__ void dpb2_kernel(const float* __restrict__ h1,
                            const float* __restrict__ w2,
                            const float* __restrict__ b2,
                            const float* __restrict__ w3,
                            const float* __restrict__ b3,
                            float* __restrict__ tab) {
  int r = threadIdx.x;                          // 256 threads, 1 block
  float acc[NHEAD];
#pragma unroll
  for (int hh = 0; hh < NHEAD; hh++) acc[hh] = b3[hh];
  for (int c2 = 0; c2 < 256; c2++) {
    float s = b2[c2];
    for (int c1 = 0; c1 < 256; c1++) s += h1[r * 256 + c1] * w2[c1 * 256 + c2];
    float sv = s / (1.f + expf(-s));
#pragma unroll
    for (int hh = 0; hh < NHEAD; hh++) acc[hh] += sv * w3[c2 * NHEAD + hh];
  }
#pragma unroll
  for (int hh = 0; hh < NHEAD; hh++) tab[r * NHEAD + hh] = acc[hh];
}

__global__ void dpb_expand_kernel(const float* __restrict__ tab,
                                  float* __restrict__ biasb) {
  int idx = blockIdx.x * 256 + threadIdx.x;     // 8*128*256 = 262144
  int hh = idx >> 15;
  int rem = idx & 32767;
  int i = rem >> 8, j = rem & 255;
  int d = abs(i + 128 - j);
  biasb[idx] = tab[d * NHEAD + hh];
}

// ---------------------------------------------------------------------------
// GEGLU: g = a1 * gelu_exact(g1), padded to FFP columns (zero pad)
// u has row stride FF2P; logical width 2730.
// ---------------------------------------------------------------------------
__global__ __launch_bounds__(256) void geglu_kernel(const f16* __restrict__ u,
                                                    f16* __restrict__ gbuf) {
  int m = blockIdx.x;
  for (int f = threadIdx.x; f < FFP; f += 256) {
    float val = 0.f;
    if (f < FF1) {
      float a1 = (float)u[(size_t)m * FF2P + f];
      float g1 = (float)u[(size_t)m * FF2P + FF1 + f];
      val = a1 * 0.5f * g1 * (1.f + erff(g1 * 0.70710678118654752f));
    }
    gbuf[(size_t)m * FFP + f] = (f16)val;
  }
}

// ---------------------------------------------------------------------------
// f16 WMMA GEMM: C[M,N] = A[M,K] (f16, lda) x Bt^T, with Bt[N][Kpad] f16
// pre-transposed weights. All of M, N multiples of 128; K multiple of 64.
// Tiles staged via GLOBAL_LOAD_ASYNC_TO_LDS_B128.
// modes: 0 -> outH f16 ; 1 -> outF = res + acc (f32) ; 2 -> outF = acc (f32)
// ---------------------------------------------------------------------------
#define BM 128
#define BN 128
#define BK 64
#define LDT 72     // halves; 144B row stride (16B aligned)

__global__ __launch_bounds__(256) void gemm_kernel(
    const f16* __restrict__ A, int lda,
    const f16* __restrict__ Bt, int ldbt,
    int N, int Kl,
    float* __restrict__ outF, f16* __restrict__ outH,
    const float* __restrict__ res, int mode) {
  __shared__ f16 As[BM * LDT];
  __shared__ f16 Bs[BN * LDT];
  int m0 = blockIdx.y * BM, n0 = blockIdx.x * BN;
  int t = threadIdx.x, wave = t >> 5, lane = t & 31;
  int wm = wave >> 2, wn = wave & 3;        // 2 x 4 wave grid
  int hi = lane >> 4, ln = lane & 15;

  int lrow = t >> 1, lseg = (t & 1) * 32;   // loader mapping: 64B per thread
  const f16* srcA0 = A + (size_t)(m0 + lrow) * lda + lseg;
  const f16* srcB0 = Bt + (size_t)(n0 + lrow) * ldbt + lseg;
  unsigned ldsA = (unsigned)(uintptr_t)(&As[lrow * LDT + lseg]);
  unsigned ldsB = (unsigned)(uintptr_t)(&Bs[lrow * LDT + lseg]);

  v8f acc[4][2];
#pragma unroll
  for (int a = 0; a < 4; a++)
#pragma unroll
    for (int b = 0; b < 2; b++) acc[a][b] = (v8f){0.f,0.f,0.f,0.f,0.f,0.f,0.f,0.f};

  for (int k0 = 0; k0 < Kl; k0 += BK) {
    // --- async DMA stage of A and B tiles (no VGPR round trip) ---
    async_copy64(ldsA, (unsigned long long)(uintptr_t)(srcA0 + k0));
    async_copy64(ldsB, (unsigned long long)(uintptr_t)(srcB0 + k0));
    async_wait0();
    __syncthreads();

#pragma unroll
    for (int ks = 0; ks < 2; ks++) {
      v16h bfrag[2];
#pragma unroll
      for (int nt = 0; nt < 2; nt++) {
        const f16* p = &Bs[(wn * 32 + nt * 16 + ln) * LDT + ks * 32 + hi * 8];
        bfrag[nt] = ld16(p, p + 16);
      }
#pragma unroll
      for (int mt = 0; mt < 4; mt++) {
        const f16* p = &As[(wm * 64 + mt * 16 + ln) * LDT + ks * 32 + hi * 8];
        v16h afrag = ld16(p, p + 16);
#pragma unroll
        for (int nt = 0; nt < 2; nt++)
          acc[mt][nt] = wmma16(afrag, bfrag[nt], acc[mt][nt]);
      }
    }
    __syncthreads();
  }

  // --- epilogue: C layout lane(r,hi) -> M = r + 8*hi, N = ln ---
#pragma unroll
  for (int mt = 0; mt < 4; mt++)
#pragma unroll
    for (int nt = 0; nt < 2; nt++)
#pragma unroll
      for (int r = 0; r < 8; r++) {
        int mg = m0 + wm * 64 + mt * 16 + r + 8 * hi;
        int ng = n0 + wn * 32 + nt * 16 + ln;
        float v = acc[mt][nt][r];
        size_t o = (size_t)mg * N + ng;
        if (mode == 0)      outH[o] = (f16)v;
        else if (mode == 1) outF[o] = res[o] + v;
        else                outF[o] = v;
      }
}

// ---------------------------------------------------------------------------
// Fused windowed attention. One block per (window, head, batch); 256 threads.
// sim = (q @ k2^T)*scale + bias, masked, softmax, o = a @ v2.
// qkv layout: [tok][0:512]=Q [512:1024]=K [1024:1536]=V, f16.
// ---------------------------------------------------------------------------
#define VT_S 264   // halves; 528B stride (16B aligned)
#define PA_S 72    // halves; 144B stride (16B aligned)

__global__ __launch_bounds__(256) void attn_kernel(
    const f16* __restrict__ qkv, const float* __restrict__ biasb,
    const float* __restrict__ maskf, f16* __restrict__ obuf) {
  const int w = blockIdx.x, head = blockIdx.y, b = blockIdx.z;
  __shared__ f16 vt[DHD * VT_S];      // v2 transposed: vt[dh][j]
  __shared__ f16 pa[WSZ * PA_S];      // prob chunk (128 rows x 64 keys)

  int t = threadIdx.x, wave = t >> 5, lane = t & 31;
  int hi = lane >> 4, ln = lane & 15;
  const int wprev = (w == 0) ? 0 : (w - 1);

  // --- stage v2^T into LDS (vector loads, b16 transpose stores) ---
  {
    int j = t;                                   // key 0..255
    int keyrow = (j < WSZ) ? (wprev * WSZ + j) : (w * WSZ + j - WSZ);
    const f16* src = qkv + ((size_t)(b * SEQ + keyrow)) * QKVN + 1024 + head * DHD;
    v8h vv[8];
#pragma unroll
    for (int c = 0; c < 8; c++) vv[c] = *(const v8h*)(src + c * 8);
#pragma unroll
    for (int c = 0; c < 8; c++)
#pragma unroll
      for (int e = 0; e < 8; e++) vt[(c * 8 + e) * VT_S + j] = vv[c][e];
  }

  // --- Q fragments (A layout), kept resident ---
  int qrow = w * WSZ + wave * 16 + ln;
  const f16* qp = qkv + ((size_t)(b * SEQ + qrow)) * QKVN + head * DHD;
  v16h aq[2];
  aq[0] = ld16(qp + hi * 8,      qp + 16 + hi * 8);
  aq[1] = ld16(qp + 32 + hi * 8, qp + 48 + hi * 8);

  // --- QK^T: wave owns rows [wave*16, wave*16+15], 16 n-tiles of keys ---
  v8f s[16];
#pragma unroll
  for (int nt = 0; nt < 16; nt++) {
    v8f acc = (v8f){0.f,0.f,0.f,0.f,0.f,0.f,0.f,0.f};
    int j = nt * 16 + ln;
    int keyrow = (j < WSZ) ? (wprev * WSZ + j) : (w * WSZ + j - WSZ);
    const f16* kp = qkv + ((size_t)(b * SEQ + keyrow)) * QKVN + 512 + head * DHD;
#pragma unroll
    for (int ks = 0; ks < 2; ks++) {
      v16h bf = ld16(kp + ks * 32 + hi * 8, kp + ks * 32 + hi * 8 + 16);
      acc = wmma16(aq[ks], bf, acc);
    }
    s[nt] = acc;
  }

  // --- scale + bias + masks ---
  const float SCALE = 0.125f;                    // 64^-0.5
  const float NEG = -3.402823466e38f;
#pragma unroll
  for (int nt = 0; nt < 16; nt++) {
    int j = nt * 16 + ln;
    int kpos = (j < WSZ) ? ((w - 1) * WSZ + j) : (w * WSZ + j - WSZ);
    int keyrow = (j < WSZ) ? (wprev * WSZ + j) : (w * WSZ + j - WSZ);
    bool jok = !(w == 0 && j < WSZ);
    float kmv = maskf[b * SEQ + keyrow];
    const float* bptr = biasb + ((size_t)head * WSZ + wave * 16) * 256 + j;
#pragma unroll
    for (int r = 0; r < 8; r++) {
      int i_loc = wave * 16 + r + 8 * hi;
      int i_abs = w * WSZ + i_loc;
      float v = s[nt][r] * SCALE + bptr[(r + 8 * hi) * 256];
      bool ok = jok && (kmv != 0.f) && (i_abs >= kpos) && (i_abs <= kpos + WSZ);
      s[nt][r] = ok ? v : NEG;
    }
  }

  // --- softmax over 256 keys per row (row lives in a 16-lane half) ---
  float mx[8], sm[8];
#pragma unroll
  for (int r = 0; r < 8; r++) {
    float m = s[0][r];
#pragma unroll
    for (int nt = 1; nt < 16; nt++) m = fmaxf(m, s[nt][r]);
    mx[r] = m;
  }
#pragma unroll
  for (int d = 1; d < 16; d <<= 1)
#pragma unroll
    for (int r = 0; r < 8; r++) mx[r] = fmaxf(mx[r], __shfl_xor(mx[r], d, 32));
#pragma unroll
  for (int r = 0; r < 8; r++) sm[r] = 0.f;
#pragma unroll
  for (int nt = 0; nt < 16; nt++)
#pragma unroll
    for (int r = 0; r < 8; r++) {
      float e = expf(s[nt][r] - mx[r]);
      s[nt][r] = e;
      sm[r] += e;
    }
#pragma unroll
  for (int d = 1; d < 16; d <<= 1)
#pragma unroll
    for (int r = 0; r < 8; r++) sm[r] += __shfl_xor(sm[r], d, 32);
#pragma unroll
  for (int r = 0; r < 8; r++) sm[r] = 1.f / sm[r];
#pragma unroll
  for (int nt = 0; nt < 16; nt++)
#pragma unroll
    for (int r = 0; r < 8; r++) s[nt][r] *= sm[r];

  // --- AV: process keys in 4 chunks of 64, restaging probs via LDS ---
  v8f o[4];
#pragma unroll
  for (int d = 0; d < 4; d++) o[d] = (v8f){0.f,0.f,0.f,0.f,0.f,0.f,0.f,0.f};

  for (int c = 0; c < 4; c++) {
    __syncthreads();   // vt ready (c==0); pa reads of prev chunk done (c>0)
#pragma unroll
    for (int nn = 0; nn < 4; nn++) {
      int nt = c * 4 + nn;
#pragma unroll
      for (int r = 0; r < 8; r++)
        pa[(wave * 16 + r + 8 * hi) * PA_S + nn * 16 + ln] = (f16)s[nt][r];
    }
    __syncthreads();
#pragma unroll
    for (int ks = 0; ks < 2; ks++) {
      const f16* ap = &pa[(wave * 16 + ln) * PA_S + ks * 32 + hi * 8];
      v16h af = ld16(ap, ap + 16);
#pragma unroll
      for (int d = 0; d < 4; d++) {
        const f16* bp = &vt[(d * 16 + ln) * VT_S + c * 64 + ks * 32 + hi * 8];
        v16h bf = ld16(bp, bp + 16);
        o[d] = wmma16(af, bf, o[d]);
      }
    }
  }

  // --- store o in (B, N, H*DH) f16 ---
#pragma unroll
  for (int d = 0; d < 4; d++)
#pragma unroll
    for (int r = 0; r < 8; r++) {
      int i_loc = wave * 16 + r + 8 * hi;
      obuf[((size_t)(b * SEQ + w * WSZ + i_loc)) * D_MODEL + head * DHD + d * 16 + ln] =
          (f16)o[d][r];
    }
}

// ---------------------------------------------------------------------------
// Host-side launch sequence
// ---------------------------------------------------------------------------
extern "C" void kernel_launch(void* const* d_in, const int* in_sizes, int n_in,
                              void* d_out, int out_size, void* d_ws, size_t ws_size,
                              hipStream_t stream) {
  (void)in_sizes; (void)n_in; (void)out_size; (void)ws_size;
  const int*   x        = (const int*)d_in[0];
  const float* emb      = (const float*)d_in[1];
  const float* ln_ag    = (const float*)d_in[2];
  const float* ln_ab    = (const float*)d_in[3];
  const float* w_qkv    = (const float*)d_in[4];
  const float* w_out    = (const float*)d_in[5];
  const float* ln_fg    = (const float*)d_in[6];
  const float* ln_fb    = (const float*)d_in[7];
  const float* w_ff1    = (const float*)d_in[8];
  const float* w_ff2    = (const float*)d_in[9];
  const float* dpb_w1   = (const float*)d_in[10];
  const float* dpb_b1   = (const float*)d_in[11];
  const float* dpb_w2   = (const float*)d_in[12];
  const float* dpb_b2   = (const float*)d_in[13];
  const float* dpb_w3   = (const float*)d_in[14];
  const float* dpb_b3   = (const float*)d_in[15];
  const float* ln_og    = (const float*)d_in[16];
  const float* ln_ob    = (const float*)d_in[17];
  const float* w_logits = (const float*)d_in[18];
  float* out = (float*)d_out;

  // workspace layout (bytes)
  char* ws = (char*)d_ws;
  float* hbuf  = (float*)(ws);                         // 16384*512 f32    33,554,432
  f16*   zbuf  = (f16*)  (ws + 33554432ull);           // 16384*512 f16    16,777,216
  f16*   qkvb  = (f16*)  (ws + 50331648ull);           // 16384*1536 f16   50,331,648
  f16*   ob    = (f16*)  (ws + 100663296ull);          // 16384*512 f16    16,777,216
  f16*   ub    = (f16*)  (ws + 117440512ull);          // 16384*2816 f16   92,274,688
  f16*   gb    = (f16*)  (ws + 209715200ull);          // 16384*1376 f16   45,088,768
  float* maskf = (float*)(ws + 254803968ull);          // 16384 f32            65,536
  float* biasb = (float*)(ws + 254869504ull);          // 8*128*256 f32     1,048,576
  float* h1    = (float*)(ws + 255918080ull);          // 256*256 f32         262,144
  float* tab   = (float*)(ws + 256180224ull);          // 256*8 f32             8,192
  f16*   wq_t  = (f16*)  (ws + 256188416ull);          // 6*1536*512 f16    9,437,184
  f16*   wo_t  = (f16*)  (ws + 265625600ull);          // 6*512*512 f16     3,145,728
  f16*   wf1_t = (f16*)  (ws + 268771328ull);          // 6*2816*512 f16   17,301,504
  f16*   wf2_t = (f16*)  (ws + 286072832ull);          // 6*512*1376 f16    8,454,144
  f16*   wlg_t = (f16*)  (ws + 294526976ull);          // 512*512 f16         524,288
                                                       // total ~295 MB

  // --- one-time weight cast/transpose (per launch) ---
  for (int l = 0; l < 6; l++) {
    wcast_kernel<<<dim3((1536 * 512 + 255) / 256), 256, 0, stream>>>(
        w_qkv + (size_t)l * 512 * QKVN, wq_t + (size_t)l * QKVN * 512,
        512, QKVN, 512, QKVN);
    wcast_kernel<<<dim3((512 * 512 + 255) / 256), 256, 0, stream>>>(
        w_out + (size_t)l * 512 * 512, wo_t + (size_t)l * 512 * 512,
        512, 512, 512, 512);
    wcast_kernel<<<dim3((FF2P * 512 + 255) / 256), 256, 0, stream>>>(
        w_ff1 + (size_t)l * 512 * FF2, wf1_t + (size_t)l * FF2P * 512,
        512, FF2, 512, FF2P);
    wcast_kernel<<<dim3((512 * FFP + 255) / 256), 256, 0, stream>>>(
        w_ff2 + (size_t)l * FF1 * 512, wf2_t + (size_t)l * 512 * FFP,
        FF1, 512, FFP, 512);
  }
  wcast_kernel<<<dim3((512 * 512 + 255) / 256), 256, 0, stream>>>(
      w_logits, wlg_t, 512, 512, 512, 512);

  embed_kernel<<<dim3(NTOK), 256, 0, stream>>>(x, emb, hbuf, maskf);
  dpb1_kernel<<<dim3(256), 256, 0, stream>>>(dpb_w1, dpb_b1, h1);
  dpb2_kernel<<<dim3(1), 256, 0, stream>>>(h1, dpb_w2, dpb_b2, dpb_w3, dpb_b3, tab);
  dpb_expand_kernel<<<dim3(1024), 256, 0, stream>>>(tab, biasb);

  for (int l = 0; l < 6; l++) {
    // attention block
    ln_cast_kernel<<<dim3(NTOK), 256, 0, stream>>>(hbuf, ln_ag + l * 512, ln_ab + l * 512, zbuf);
    gemm_kernel<<<dim3(QKVN / BN, NTOK / BM), 256, 0, stream>>>(
        zbuf, D_MODEL, wq_t + (size_t)l * QKVN * 512, 512,
        QKVN, 512, nullptr, qkvb, nullptr, 0);
    attn_kernel<<<dim3(NWIN, NHEAD, 8), 256, 0, stream>>>(qkvb, biasb, maskf, ob);
    gemm_kernel<<<dim3(D_MODEL / BN, NTOK / BM), 256, 0, stream>>>(
        ob, D_MODEL, wo_t + (size_t)l * 512 * 512, 512,
        D_MODEL, 512, hbuf, nullptr, hbuf, 1);
    // GEGLU feed-forward block
    ln_cast_kernel<<<dim3(NTOK), 256, 0, stream>>>(hbuf, ln_fg + l * 512, ln_fb + l * 512, zbuf);
    gemm_kernel<<<dim3(FF2P / BN, NTOK / BM), 256, 0, stream>>>(
        zbuf, D_MODEL, wf1_t + (size_t)l * FF2P * 512, 512,
        FF2P, 512, nullptr, ub, nullptr, 0);
    geglu_kernel<<<dim3(NTOK), 256, 0, stream>>>(ub, gb);
    gemm_kernel<<<dim3(D_MODEL / BN, NTOK / BM), 256, 0, stream>>>(
        gb, FFP, wf2_t + (size_t)l * 512 * FFP, FFP,
        D_MODEL, FFP, hbuf, nullptr, hbuf, 1);
  }

  // final LN + logits projection (f32 output)
  ln_cast_kernel<<<dim3(NTOK), 256, 0, stream>>>(hbuf, ln_og, ln_ob, zbuf);
  gemm_kernel<<<dim3(D_MODEL / BN, NTOK / BM), 256, 0, stream>>>(
      zbuf, D_MODEL, wlg_t, 512,
      D_MODEL, 512, out, nullptr, nullptr, 2);
}
